// SignatureCrossAttention_81037442941451
// MI455X (gfx1250) — matrix operand order repaired and checked by reading, not compiled, not run on hardware
//
#include <hip/hip_runtime.h>
#include <hip/hip_bf16.h>
#include <stdint.h>

// ---------------------------------------------------------------------------
// SignatureCrossAttention for MI455X (gfx1250, wave32, WMMA bf16 + TDM)
//   Stage 1: per-path depth-2 signature via v_wmma_f32_16x16x32_bf16
//   Stage 2: fused dual GEMM (sq@WqT, (sk*sv)@WkvT); A-side tiles staged into
//            LDS by the Tensor Data Mover (tensor_load_to_lds, TENSORcnt),
//            B-side (fp32 weights) converted to bf16 on the fly.
// ---------------------------------------------------------------------------

typedef __bf16 bf16_t;
typedef bf16_t v16bf __attribute__((ext_vector_type(16)));
typedef float  v8f   __attribute__((ext_vector_type(8)));
typedef unsigned int v8u __attribute__((ext_vector_type(8)));
typedef unsigned int v4u __attribute__((ext_vector_type(4)));
typedef int v8i __attribute__((ext_vector_type(8)));
typedef int v4i __attribute__((ext_vector_type(4)));
typedef unsigned short u16;

static constexpr int Bsz = 32, Lc = 128, Hh = 16, Es = 96;
static constexpr int NPATH = Bsz * Hh;              // 512 paths
static constexpr int SIG   = Lc + Lc * Lc;          // 16512
static constexpr int OUTD  = 768;
static constexpr int HE    = Hh * Es;               // 1536
static constexpr size_t LHE = (size_t)Lc * Hh * Es; // 196608
static constexpr int RS  = 104;  // sig u/dx LDS row stride (u16), 8-mult => 16B rows
static constexpr int CRS = 40;   // gemm B chunk LDS row stride (u16)
static constexpr int ARS = 136;  // gemm A chunk LDS row stride (u16): 256B + 16B pad
static constexpr int KCH = 128;  // A chunk K depth (bf16) per TDM stage
static constexpr int NCHUNK = SIG / KCH;            // 129

#if defined(__has_builtin)
#  if __has_builtin(__builtin_amdgcn_tensor_load_to_lds)
#    define USE_TDM 1
#  endif
#endif
#ifndef USE_TDM
#  define USE_TDM 0
#endif

__device__ __forceinline__ u16 f2bf(float f) {  // RNE float->bf16
  unsigned int u = __builtin_bit_cast(unsigned int, f);
  u += 0x7FFFu + ((u >> 16) & 1u);
  return (u16)(u >> 16);
}

__device__ __forceinline__ v8f vzero() {
  v8f z;
#pragma unroll
  for (int i = 0; i < 8; ++i) z[i] = 0.f;
  return z;
}

// A fragment 16x32 bf16 (MxK). Lanes 0-15: M=lane, V0..3=K{0..7}, V4..7=K{16..23};
// lanes 16-31: K+8. rs must be a multiple of 8 u16 (16B rows) -> 2x ds_load_b128.
__device__ __forceinline__ v16bf frag_A(const u16* base, int rs, int c0, int kb, int lane) {
  const int m = lane & 15, hi = lane >> 4;
  const u16* p = base + (c0 + m) * rs + kb + hi * 8;
  const uint4 a = *(const uint4*)p;
  const uint4 b = *(const uint4*)(p + 16);
  v8u u;
  u[0] = a.x; u[1] = a.y; u[2] = a.z; u[3] = a.w;
  u[4] = b.x; u[5] = b.y; u[6] = b.z; u[7] = b.w;
  return __builtin_bit_cast(v16bf, u);
}

// B fragment 32x16 bf16 (KxN), storage row-major [n][k]. Lanes 0-15: N=lane,
// V0..3=K{0..7}, V4..7=K{16..23}; lanes 16-31: K+8. 2x ds_load_b128.
__device__ __forceinline__ v16bf frag_B(const u16* base, int rs, int c0, int kb, int lane) {
  const int n = lane & 15, hi = lane >> 4;
  const u16* p = base + (c0 + n) * rs + kb + hi * 8;
  const uint4 a = *(const uint4*)p;
  const uint4 b = *(const uint4*)(p + 16);
  v8u u;
  u[0] = a.x; u[1] = a.y; u[2] = a.z; u[3] = a.w;
  u[4] = b.x; u[5] = b.y; u[6] = b.z; u[7] = b.w;
  return __builtin_bit_cast(v16bf, u);
}

#if USE_TDM
// Issue a TDM 2D tile load: 64 rows x 256B from a bf16 matrix with row pitch
// SIG u16, into LDS with 16B padding per 256B row (=> ARS u16 row stride).
// D# per CDNA5 ISA 8.3/8.4: group0 = {count|flags, lds_addr, global_addr, type},
// group1 = {mask/size/pad, dims/strides}. data_size=4B; pad every 64 dwords by 4.
__device__ __forceinline__ void tdm_load_2d(unsigned lds_off, const u16* gptr) {
  const unsigned long long ga = (unsigned long long)(size_t)gptr;
  v4u g0;
  g0[0] = 1u;                                        // count=1, user mode
  g0[1] = lds_off;                                   // lds_addr (bytes)
  g0[2] = (unsigned)ga;                              // global_addr[31:0]
  g0[3] = (unsigned)((ga >> 32) & 0x01FFFFFFull) | 0x80000000u;  // [56:32] | type=2
  v8i g1;
  g1[0] = (int)((2u << 16) | (1u << 20) | (5u << 22) | (3u << 25));
  //        data_size=4B    pad_en       pad_int=64dw  pad_amt=4dw ; wg_mask=0
  g1[1] = (int)((8256u & 0xFFFFu) << 16);            // td0[15:0]  (8256 dw/row)
  g1[2] = (int)((8256u >> 16) | (64u << 16));        // td0[31:16] | td1[15:0]=64
  g1[3] = (int)(64u << 16);                          // td1[31:16] | tile_dim0=64dw
  g1[4] = (int)(64u);                                // tile_dim1=64 rows, tile_dim2=0
  g1[5] = (int)(8256u);                              // tensor_dim0_stride[31:0]
  g1[6] = 0;                                         // stride hi | td1_stride lo
  g1[7] = 0;
  const v4i z4 = {0, 0, 0, 0};
#if defined(__clang_major__) && __clang_major__ >= 23
  const v8i z8 = {0, 0, 0, 0, 0, 0, 0, 0};
  __builtin_amdgcn_tensor_load_to_lds(g0, g1, z4, z4, z8, 0);
#else
  __builtin_amdgcn_tensor_load_to_lds(g0, g1, z4, z4, 0);
#endif
}
#endif

// ---------------------------------------------------------------------------
// Kernel 1: depth-2 signatures. grid = (512, 2); block = 256 (8 waves).
//   blockIdx.y==0 : sq = sig(q) ; blockIdx.y==1 : skv = sig(k)*sig(v)   (bf16)
// lvl2 (128x128) = uT(128x96) @ dx(96x128), K padded to 96.
// ---------------------------------------------------------------------------
__global__ __launch_bounds__(256) void sig_kernel(
    const float* __restrict__ q, const float* __restrict__ k,
    const float* __restrict__ v,
    u16* __restrict__ sq, u16* __restrict__ skv) {
  extern __shared__ char smem[];
  float* xld = (float*)smem;                       // 128 x 96 f32
  u16*   uld = (u16*)(smem + Lc * Es * 4);         // 128 x RS bf16
  u16*   dld = uld + Lc * RS;                      // 128 x RS bf16

  const int n    = blockIdx.x;
  const int mode = blockIdx.y;
  const int tid  = threadIdx.x;
  const int lane = tid & 31;
  const int wv   = tid >> 5;
  const size_t nbase = (size_t)(n >> 4) * LHE + (size_t)(n & 15) * Es;

  v8f accA[8], accB[8];

  auto compute = [&](const float* __restrict__ x, v8f acc[8]) {
    __syncthreads();
    for (int idx = tid; idx < Lc * (Es / 4); idx += 256) {
      const int c  = idx / (Es / 4);
      const int e4 = (idx - c * (Es / 4)) * 4;
      const float4 t = *(const float4*)(x + (size_t)c * HE + e4);
      *(float4*)(xld + c * Es + e4) = t;
    }
    __syncthreads();
    for (int idx = tid; idx < Lc * Es; idx += 256) {
      const int c = idx / Es;
      const int s = idx - c * Es;
      const float xs = xld[c * Es + s];
      float uv = 0.f, dv = 0.f;
      if (s < Es - 1) {
        const float xn = xld[c * Es + s + 1];
        const float x0 = xld[c * Es];
        dv = xn - xs;
        uv = 0.5f * (xs + xn) - x0;
      }
      uld[c * RS + s] = f2bf(uv);
      dld[c * RS + s] = f2bf(dv);
    }
    __syncthreads();
#pragma unroll
    for (int b = 0; b < 8; ++b) acc[b] = vzero();
    for (int kb = 0; kb < Es; kb += 32) {
      const v16bf af = frag_A(uld, RS, wv * 16, kb, lane);
#pragma unroll
      for (int b = 0; b < 8; ++b) {
        const v16bf bfv = frag_B(dld, RS, b * 16, kb, lane);
        acc[b] = __builtin_amdgcn_wmma_f32_16x16x32_bf16(
            false, af, false, bfv, (short)0, acc[b], false, false);
      }
    }
  };

  u16* orow;
  if (mode == 0) {
    compute(q + nbase, accA);
    orow = sq + (size_t)n * SIG;
    if (tid < Lc) {
      const float l1 = xld[tid * Es + (Es - 1)] - xld[tid * Es];
      orow[tid] = f2bf(l1);
    }
  } else {
    compute(k + nbase, accA);
    compute(v + nbase, accB);
    orow = skv + (size_t)n * SIG;
    if (tid < Lc) {
      const float* kp = k + nbase + (size_t)tid * HE;
      const float l1k = kp[Es - 1] - kp[0];
      const float l1v = xld[tid * Es + (Es - 1)] - xld[tid * Es];
      orow[tid] = f2bf(l1k * l1v);
    }
  }
  const int hi = lane >> 4, nn = lane & 15;
#pragma unroll
  for (int b = 0; b < 8; ++b) {
#pragma unroll
    for (int r = 0; r < 8; ++r) {
      const int a  = wv * 16 + r + hi * 8;
      const int cc = b * 16 + nn;
      float val = accA[b][r];
      if (mode != 0) val *= accB[b][r];
      orow[Lc + a * Lc + cc] = f2bf(val);
    }
  }
}

// ---------------------------------------------------------------------------
// Kernel 2: fused dual GEMM + epilogue product.
//   M=512 x N=768 x K=16512, block tile 64x128, grid 8x6, 8 waves.
//   A (bf16 sigs): TDM double-buffered LDS staging, 129 chunks of K=128.
//   B (fp32 weights): register->LDS pipeline with on-the-fly bf16 convert.
// ---------------------------------------------------------------------------
__global__ __launch_bounds__(256) void gemm_kernel(
    const u16* __restrict__ sqm, const u16* __restrict__ skvm,
    const float* __restrict__ Wq, const float* __restrict__ Wkv,
    const float* __restrict__ bq, const float* __restrict__ bkv,
    float* __restrict__ out) {
  extern __shared__ char smem[];
  u16* aq_s[2]; u16* ak_s[2];
  aq_s[0] = (u16*)smem;
  aq_s[1] = aq_s[0] + 64 * ARS;
  ak_s[0] = aq_s[1] + 64 * ARS;
  ak_s[1] = ak_s[0] + 64 * ARS;
  u16* bq_s = ak_s[1] + 64 * ARS;    // 128 x CRS
  u16* bk_s = bq_s + 128 * CRS;

  const int tid  = threadIdx.x;
  const int lane = tid & 31, wv = tid >> 5;
  const int m0 = blockIdx.x * 64;
  const int n0 = blockIdx.y * 128;

  const u16* gAq = sqm  + (size_t)m0 * SIG;
  const u16* gAk = skvm + (size_t)m0 * SIG;

  const int rowB = tid >> 3, partB = tid & 7;      // 128 rows x 8 float4/row
  const float* gBq = Wq  + (size_t)(n0 + rowB) * SIG + partB * 4;
  const float* gBk = Wkv + (size_t)(n0 + rowB) * SIG + partB * 4;

  uint2 rBq[4], rBk[4];
  auto gloadB = [&](int kb) {
#pragma unroll
    for (int i = 0; i < 4; ++i) {
      const float4 fq = *(const float4*)(gBq + (size_t)32 * i * SIG + kb);
      const float4 fk = *(const float4*)(gBk + (size_t)32 * i * SIG + kb);
      rBq[i].x = (unsigned)f2bf(fq.x) | ((unsigned)f2bf(fq.y) << 16);
      rBq[i].y = (unsigned)f2bf(fq.z) | ((unsigned)f2bf(fq.w) << 16);
      rBk[i].x = (unsigned)f2bf(fk.x) | ((unsigned)f2bf(fk.y) << 16);
      rBk[i].y = (unsigned)f2bf(fk.z) | ((unsigned)f2bf(fk.w) << 16);
    }
  };

  v8f accq[4], acck[4];
#pragma unroll
  for (int t = 0; t < 4; ++t) { accq[t] = vzero(); acck[t] = vzero(); }

  const int mt = (wv & 3) * 16;
  const int nh = (wv >> 2) * 64;

#if USE_TDM
  if (wv == 0) {  // one wave issues DMA; TDM ignores EXEC but issues per wave
    tdm_load_2d((unsigned)(size_t)aq_s[0], gAq);
    tdm_load_2d((unsigned)(size_t)ak_s[0], gAk);
    tdm_load_2d((unsigned)(size_t)aq_s[1], gAq + KCH);
    tdm_load_2d((unsigned)(size_t)ak_s[1], gAk + KCH);
  }
#endif
  gloadB(0);

  for (int oc = 0; oc < NCHUNK; ++oc) {
    u16* aqb = aq_s[oc & 1];
    u16* akb = ak_s[oc & 1];
#if USE_TDM
    if (wv == 0) {
      if (oc < NCHUNK - 1) __builtin_amdgcn_s_wait_tensorcnt(2);  // pair oc done
      else                 __builtin_amdgcn_s_wait_tensorcnt(0);
    }
#else
    for (int idx = tid; idx < 64 * (KCH / 8); idx += 256) {  // fallback copy
      const int r = idx >> 4, p4 = idx & 15;
      *(uint4*)(aqb + r * ARS + p4 * 8) =
          *(const uint4*)(gAq + (size_t)r * SIG + oc * KCH + p4 * 8);
      *(uint4*)(akb + r * ARS + p4 * 8) =
          *(const uint4*)(gAk + (size_t)r * SIG + oc * KCH + p4 * 8);
    }
#endif
#pragma unroll
    for (int j = 0; j < 4; ++j) {
      const int kglob = oc * KCH + j * 32;
      __syncthreads();  // also releases waves after wave0's tensorcnt wait
#pragma unroll
      for (int i = 0; i < 4; ++i) {
        *(uint2*)(bq_s + (rowB + 32 * i) * CRS + partB * 4) = rBq[i];
        *(uint2*)(bk_s + (rowB + 32 * i) * CRS + partB * 4) = rBk[i];
      }
      __syncthreads();
      if (kglob + 32 < SIG) {
        gloadB(kglob + 32);
        __builtin_prefetch(gBq + (size_t)(kglob + 96), 0, 1);  // global_prefetch
        __builtin_prefetch(gBk + (size_t)(kglob + 96), 0, 1);
      }
      const v16bf afq = frag_A(aqb, ARS, mt, j * 32, lane);
      const v16bf afk = frag_A(akb, ARS, mt, j * 32, lane);
#pragma unroll
      for (int t = 0; t < 4; ++t) {
        const v16bf bfq = frag_B(bq_s, CRS, nh + t * 16, 0, lane);
        accq[t] = __builtin_amdgcn_wmma_f32_16x16x32_bf16(
            false, afq, false, bfq, (short)0, accq[t], false, false);
        const v16bf bfk = frag_B(bk_s, CRS, nh + t * 16, 0, lane);
        acck[t] = __builtin_amdgcn_wmma_f32_16x16x32_bf16(
            false, afk, false, bfk, (short)0, acck[t], false, false);
      }
    }
    __syncthreads();  // all reads of A[cur] done before buffer reuse
#if USE_TDM
    if (oc + 2 < NCHUNK && wv == 0) {
      tdm_load_2d((unsigned)(size_t)aqb, gAq + (size_t)(oc + 2) * KCH);
      tdm_load_2d((unsigned)(size_t)akb, gAk + (size_t)(oc + 2) * KCH);
    }
#endif
  }

  // epilogue: (qt+bq)*(kvt+bkv), permuted to out[b][j][h][e], o = e*8+j
  const int hi = lane >> 4, nn = lane & 15;
#pragma unroll
  for (int t = 0; t < 4; ++t) {
    const int o = n0 + nh + t * 16 + nn;
    const float biasq = bq[o], biask = bkv[o];
    const int e = o >> 3, j = o & 7;
#pragma unroll
    for (int r = 0; r < 8; ++r) {
      const int nrow = m0 + mt + r + hi * 8;
      const int bb = nrow >> 4, hh = nrow & 15;
      const float qv = accq[t][r] + biasq;
      const float kv = acck[t][r] + biask;
      out[(((size_t)bb * 8 + j) * 16 + hh) * 96 + e] = qv * kv;
    }
  }
}

// ---------------------------------------------------------------------------
extern "C" void kernel_launch(void* const* d_in, const int* in_sizes, int n_in,
                              void* d_out, int out_size, void* d_ws, size_t ws_size,
                              hipStream_t stream) {
  (void)in_sizes; (void)n_in; (void)out_size; (void)ws_size;
  const float* q   = (const float*)d_in[0];
  const float* k   = (const float*)d_in[1];
  const float* v   = (const float*)d_in[2];
  // d_in[3] = mask (unused by the reference math)
  const float* Wq  = (const float*)d_in[4];
  const float* bq  = (const float*)d_in[5];
  const float* Wkv = (const float*)d_in[6];
  const float* bkv = (const float*)d_in[7];
  float* out = (float*)d_out;

  const size_t SN = (size_t)NPATH * SIG;
  u16* sqb  = (u16*)d_ws;        // bf16 sq    (16.9 MB)
  u16* skvb = sqb + SN;          // bf16 sk*sv (16.9 MB)

  {
    dim3 grid(NPATH, 2);
    const size_t lds = (size_t)Lc * Es * 4 + 2 * (size_t)Lc * RS * 2;  // 102400 B
    sig_kernel<<<grid, 256, lds, stream>>>(q, k, v, sqb, skvb);
  }
  {
    dim3 grid(512 / 64, OUTD / 128);  // 8 x 6
    const size_t lds = 4 * (size_t)64 * ARS * 2 + 2 * (size_t)128 * CRS * 2;  // 90112 B
    gemm_kernel<<<grid, 256, lds, stream>>>(sqb, skvb, Wq, Wkv, bq, bkv, out);
  }
}
